// CausalSelfAttention_49847390437478
// MI455X (gfx1250) — compile-verified, matrix operand order checked by reading, not run on hardware
//
#include <hip/hip_runtime.h>
#include <hip/hip_bf16.h>

// ---------------------------------------------------------------------------
// CDNA5 (gfx1250) causal self-attention forward, bf16 WMMA pipeline with
// async global->LDS staging (ASYNCcnt) and double-buffered GEMM tiles.
// ---------------------------------------------------------------------------

typedef __attribute__((ext_vector_type(16))) __bf16 v16bf;
typedef __attribute__((ext_vector_type(8)))  float  v8f;
typedef __attribute__((ext_vector_type(4)))  int    v4i;

union Frag16 { v16bf v; uint4 q[2]; };

// ---- async global->LDS (CDNA5): guarded so the file compiles either way ---
#if defined(__has_builtin)
#if __has_builtin(__builtin_amdgcn_global_load_async_to_lds_b128) && \
    __has_builtin(__builtin_amdgcn_s_wait_asynccnt)
#define HAVE_ASYNC 1
#endif
#endif
#ifndef HAVE_ASYNC
#define HAVE_ASYNC 0
#endif

#if HAVE_ASYNC
typedef __attribute__((address_space(1))) v4i gv4i;  // global b128 payload
typedef __attribute__((address_space(3))) v4i lv4i;  // LDS b128 payload
__device__ __forceinline__ void async_cp16(const void* g, void* l) {
  v4i* gt = (v4i*)(void*)g;   // typed, generic
  v4i* lt = (v4i*)l;
  __builtin_amdgcn_global_load_async_to_lds_b128((gv4i*)gt, (lv4i*)lt, 0, 0);
}
#define ASYNC_WAIT(n) __builtin_amdgcn_s_wait_asynccnt(n)
#else
__device__ __forceinline__ void async_cp16(const void* g, void* l) {
  *(uint4*)l = *(const uint4*)g;
}
#define ASYNC_WAIT(n)
#endif

__device__ __forceinline__ unsigned short f2bfu(float f) {
  unsigned u = __builtin_bit_cast(unsigned, f);
  u += 0x7FFFu + ((u >> 16) & 1u);          // round-to-nearest-even
  return (unsigned short)(u >> 16);
}
__device__ __forceinline__ float bfu2f(unsigned short h) {
  unsigned u = ((unsigned)h) << 16;
  return __builtin_bit_cast(float, u);
}
__device__ __forceinline__ v8f vzero() {
  v8f z = {0.f, 0.f, 0.f, 0.f, 0.f, 0.f, 0.f, 0.f};
  return z;
}

// Load one 16x32 bf16 A/B-operand fragment: per the CDNA5 ISA 16-bit operand
// layout, lanes 0-15 hold K {0..7, 16..23}, lanes 16-31 hold K {8..15, 24..31}
// of their row. Two aligned 16B reads per lane.
__device__ __forceinline__ v16bf ld_frag(const unsigned short* p, int khalf) {
  Frag16 f;
  f.q[0] = *(const uint4*)(p + khalf * 8);
  f.q[1] = *(const uint4*)(p + 16 + khalf * 8);
  return f.v;
}

__device__ __forceinline__ v8f wmma_bf16(v16bf a, v16bf b, v8f c) {
  return __builtin_amdgcn_wmma_f32_16x16x32_bf16(
      /*neg_a=*/false, a, /*neg_b=*/false, b,
      /*c_mod=*/(short)0, c, /*reuse_a=*/false, /*reuse_b=*/false);
}

// ---------------------------------------------------------------------------
// fp32 -> bf16 conversion
// ---------------------------------------------------------------------------
__global__ __launch_bounds__(256) void cvt_f32_bf16(const float* __restrict__ in,
                                                    unsigned short* __restrict__ out,
                                                    int n) {
  int i = blockIdx.x * 256 + threadIdx.x;
  if (i < n) out[i] = f2bfu(in[i]);
}

// ---------------------------------------------------------------------------
// Tiled bf16 GEMM: C[M,N] = A[M,K] * Bw[N,K]^T  (both row-major, K contiguous)
// Block 128x128, BK=32, double-buffered LDS fed by async global->LDS copies.
// 8 waves, each wave computes a 64x32 tile (4x2 WMMA). fp32 or bf16 output.
// ---------------------------------------------------------------------------
#define GPITCH 40  // bf16 elements per LDS row (32 + 8 pad, keeps 16B align)

__global__ __launch_bounds__(256) void gemm_bf16_kernel(
    const unsigned short* __restrict__ A, const unsigned short* __restrict__ Bw,
    float* __restrict__ Cf, unsigned short* __restrict__ Cb,
    int M, int N, int K) {
  __shared__ unsigned short sA[2][128 * GPITCH];
  __shared__ unsigned short sB[2][128 * GPITCH];

  const int tid  = threadIdx.x;
  const int lane = tid & 31, wave = tid >> 5;
  const int khalf = lane >> 4, rl = lane & 15;
  const int wm = wave >> 2, wn = wave & 3;          // 2 x 4 wave grid
  const int m0 = blockIdx.y * 128, n0 = blockIdx.x * 128;

  v8f acc[4][2];
#pragma unroll
  for (int i = 0; i < 4; ++i)
#pragma unroll
    for (int j = 0; j < 2; ++j) acc[i][j] = vzero();

  const int lr = tid >> 2;        // 0..63 (row within half-tile)
  const int lc = (tid & 3) * 8;   // 0,8,16,24 (K offset)
  const unsigned short* gA = &A[(size_t)(m0 + lr) * K + lc];
  const unsigned short* gB = &Bw[(size_t)(n0 + lr) * K + lc];

  // 4 async 16B copies per thread per tile (2 for A, 2 for B).
  auto issue_tile = [&](int buf, int k0) {
#pragma unroll
    for (int i = 0; i < 2; ++i) {
      async_cp16(gA + (size_t)(i * 64) * K + k0,
                 &sA[buf][(i * 64 + lr) * GPITCH + lc]);
      async_cp16(gB + (size_t)(i * 64) * K + k0,
                 &sB[buf][(i * 64 + lr) * GPITCH + lc]);
    }
  };

  const int ntiles = K >> 5;
  issue_tile(0, 0);

  for (int kt = 0; kt < ntiles; ++kt) {
    const int cur = kt & 1;
    if (kt + 1 < ntiles) {
      issue_tile(cur ^ 1, (kt + 1) * 32);  // overlap next copy with this tile
      ASYNC_WAIT(4);                       // in-order: current tile landed
    } else {
      ASYNC_WAIT(0);
    }
    __syncthreads();

    v16bf af[4], bfv[2];
#pragma unroll
    for (int i = 0; i < 4; ++i)
      af[i] = ld_frag(&sA[cur][(wm * 64 + i * 16 + rl) * GPITCH], khalf);
#pragma unroll
    for (int j = 0; j < 2; ++j)
      bfv[j] = ld_frag(&sB[cur][(wn * 32 + j * 16 + rl) * GPITCH], khalf);

#pragma unroll
    for (int i = 0; i < 4; ++i)
#pragma unroll
      for (int j = 0; j < 2; ++j) acc[i][j] = wmma_bf16(af[i], bfv[j], acc[i][j]);
    __syncthreads();  // protect the buffer being refilled next iteration
  }

  // C layout: lanes 0-15 -> rows r, lanes 16-31 -> rows r+8; lane%16 = column.
#pragma unroll
  for (int i = 0; i < 4; ++i) {
#pragma unroll
    for (int j = 0; j < 2; ++j) {
#pragma unroll
      for (int r = 0; r < 8; ++r) {
        int m = m0 + wm * 64 + i * 16 + khalf * 8 + r;
        int n = n0 + wn * 32 + j * 16 + rl;
        float val = acc[i][j][r];
        if (Cf) Cf[(size_t)m * N + n] = val;
        else    Cb[(size_t)m * N + n] = f2bfu(val);
      }
    }
  }
}

// ---------------------------------------------------------------------------
// RoPE + QKV scatter.  idx enumerates (b,h,t,d) in output order.
// Writes bf16 q (pre-scaled by 1/sqrt(hd)), bf16 k, bf16 v into [B,H,T,hd]
// plus the fp32 k, v outputs required by the reference tuple.
// ---------------------------------------------------------------------------
__global__ __launch_bounds__(256) void rope_kernel(
    const unsigned short* __restrict__ qkvb,
    unsigned short* __restrict__ qb, unsigned short* __restrict__ kb,
    unsigned short* __restrict__ vb,
    float* __restrict__ kout, float* __restrict__ vout) {
  unsigned idx = blockIdx.x * 256 + threadIdx.x;   // < 4*16*2048*128
  int d = idx & 127;
  int t = (idx >> 7) & 2047;
  int h = (idx >> 18) & 15;
  int b = idx >> 22;

  size_t row = (size_t)(b * 2048 + t) * 6144;
  int col = h * 128 + d;
  float q = bfu2f(qkvb[row + col]);
  float k = bfu2f(qkvb[row + 2048 + col]);
  float v = bfu2f(qkvb[row + 4096 + col]);

  int   dp   = (d < 64) ? d + 64 : d - 64;
  float sign = (d < 64) ? -1.f : 1.f;
  float qp = bfu2f(qkvb[row + h * 128 + dp]);
  float kp = bfu2f(qkvb[row + 2048 + h * 128 + dp]);

  // inv_freq = 10000^(-(d%64)/64)
  float inv = __expf(-(float)(d & 63) * (9.210340371976184f / 64.0f));
  float ang = (float)t * inv;
  float sn = __sinf(ang), cs = __cosf(ang);

  float qr = q * cs + sign * qp * sn;
  float kr = k * cs + sign * kp * sn;

  qb[idx]  = f2bfu(qr * 0.08838834764831845f);  // fold 1/sqrt(128) into Q
  kb[idx]  = f2bfu(kr);
  vb[idx]  = f2bfu(v);
  kout[idx] = kr;
  vout[idx] = v;
}

// ---------------------------------------------------------------------------
// Flash attention (causal, online softmax).  Block = (b, h, 128 queries),
// 8 waves x 16 query rows.  Key tiles of 32.  All matmuls via bf16 WMMA.
// K tile staged via async global->LDS; V tile transposed through VGPRs.
// Writes O as bf16 [B,T,C] ready for the output projection GEMM.
// ---------------------------------------------------------------------------
__global__ __launch_bounds__(256) void flash_kernel(
    const unsigned short* __restrict__ Qm, const unsigned short* __restrict__ Km,
    const unsigned short* __restrict__ Vm, unsigned short* __restrict__ Om) {
  __shared__ unsigned short sK[32 * 136];   // [key][dim], padded
  __shared__ unsigned short sV[128 * 40];   // transposed: [dim][key], padded
  __shared__ unsigned short sP[8 * 16 * 40];// per-wave P staging

  const int T = 2048;
  const int tid  = threadIdx.x;
  const int lane = tid & 31, wave = tid >> 5;
  const int khalf = lane >> 4, rl = lane & 15;
  const int q0 = blockIdx.x * 128;
  const size_t head = ((size_t)blockIdx.z * 16 + blockIdx.y) * (size_t)T * 128;

  // Q fragments for this wave's 16 rows: 4 chained K-steps covering hd=128.
  v16bf qf[4];
  {
    const unsigned short* qp = Qm + head + (size_t)(q0 + wave * 16 + rl) * 128;
#pragma unroll
    for (int c = 0; c < 4; ++c) {
      Frag16 f;
      f.q[0] = *(const uint4*)(qp + c * 32 + khalf * 8);
      f.q[1] = *(const uint4*)(qp + c * 32 + 16 + khalf * 8);
      qf[c] = f.v;
    }
  }

  v8f o[8];
#pragma unroll
  for (int f = 0; f < 8; ++f) o[f] = vzero();
  float mrow[8], lrow[8];
#pragma unroll
  for (int r = 0; r < 8; ++r) { mrow[r] = -1e30f; lrow[r] = 0.f; }

  const int nkt = q0 / 32 + 4;  // causal: keys [0, q0+128)
  for (int kt = 0; kt < nkt; ++kt) {
    const int k0 = kt * 32;
    {
      // K tile: async global->LDS, [32 keys][128 dims] in 16B chunks.
      int key = tid >> 3, dg = (tid & 7) * 16;
      const unsigned short* kp = Km + head + (size_t)(k0 + key) * 128 + dg;
      async_cp16(kp,     &sK[key * 136 + dg]);
      async_cp16(kp + 8, &sK[key * 136 + dg + 8]);
      // V tile transposed into LDS: [dim][key] (overlaps the async K copy).
      int vkey = tid & 31, vdg = (tid >> 5) * 16;
      const unsigned short* vp = Vm + head + (size_t)(k0 + vkey) * 128 + vdg;
      union { uint4 q[2]; unsigned short u[16]; } tv;
      tv.q[0] = *(const uint4*)(vp);
      tv.q[1] = *(const uint4*)(vp + 8);
      if (kt + 1 < nkt) __builtin_prefetch(vp + 4096, 0, 1);  // next V tile
#pragma unroll
      for (int dd = 0; dd < 16; ++dd) sV[(vdg + dd) * 40 + vkey] = tv.u[dd];
      ASYNC_WAIT(0);
    }
    __syncthreads();

    // S = Q * K^T : two 16-key fragments, 4 chained WMMAs over hd each.
    v8f s[2];
    s[0] = vzero(); s[1] = vzero();
#pragma unroll
    for (int j = 0; j < 2; ++j)
#pragma unroll
      for (int c = 0; c < 4; ++c)
        s[j] = wmma_bf16(qf[c], ld_frag(&sK[(j * 16 + rl) * 136 + c * 32], khalf), s[j]);

    // Causal mask (C layout: row = rowBase + r, col = k0 + rl + 16j).
    const int rowBase = q0 + wave * 16 + khalf * 8;
    if (k0 + 31 > rowBase) {
#pragma unroll
      for (int j = 0; j < 2; ++j) {
        int key = k0 + rl + j * 16;
#pragma unroll
        for (int r = 0; r < 8; ++r)
          if (key > rowBase + r) s[j][r] = -1e30f;
      }
    }

    // Online softmax: reductions stay inside 16-lane halves (rows differ
    // between halves in the WMMA C layout).
#pragma unroll
    for (int r = 0; r < 8; ++r) {
      float mx = fmaxf(s[0][r], s[1][r]);
      mx = fmaxf(mx, __shfl_xor(mx, 1, 32));
      mx = fmaxf(mx, __shfl_xor(mx, 2, 32));
      mx = fmaxf(mx, __shfl_xor(mx, 4, 32));
      mx = fmaxf(mx, __shfl_xor(mx, 8, 32));
      float mnew  = fmaxf(mrow[r], mx);
      float alpha = __expf(mrow[r] - mnew);
      mrow[r] = mnew;
      float p0 = __expf(s[0][r] - mnew);
      float p1 = __expf(s[1][r] - mnew);
      s[0][r] = p0; s[1][r] = p1;
      float sum = p0 + p1;
      sum += __shfl_xor(sum, 1, 32);
      sum += __shfl_xor(sum, 2, 32);
      sum += __shfl_xor(sum, 4, 32);
      sum += __shfl_xor(sum, 8, 32);
      lrow[r] = lrow[r] * alpha + sum;
#pragma unroll
      for (int f = 0; f < 8; ++f) o[f][r] *= alpha;
    }

    // Re-shape P from C layout to A layout through per-wave LDS staging.
    unsigned short* pb = &sP[wave * 16 * 40];
#pragma unroll
    for (int j = 0; j < 2; ++j)
#pragma unroll
      for (int r = 0; r < 8; ++r)
        pb[(khalf * 8 + r) * 40 + j * 16 + rl] = f2bfu(s[j][r]);
    v16bf pf = ld_frag(&pb[rl * 40], khalf);

    // O += P * V : 8 output fragments covering hd=128.
#pragma unroll
    for (int f = 0; f < 8; ++f)
      o[f] = wmma_bf16(pf, ld_frag(&sV[(f * 16 + rl) * 40], khalf), o[f]);
    __syncthreads();
  }

  // Normalize and store O as bf16 [B, T, H*hd].
#pragma unroll
  for (int f = 0; f < 8; ++f) {
#pragma unroll
    for (int r = 0; r < 8; ++r) {
      float val = o[f][r] / lrow[r];
      int trow = q0 + wave * 16 + khalf * 8 + r;
      int dcol = f * 16 + rl;
      Om[((size_t)blockIdx.z * T + trow) * 2048 + blockIdx.y * 128 + dcol] =
          f2bfu(val);
    }
  }
}

// ---------------------------------------------------------------------------
// Host-side launch sequence.
// ---------------------------------------------------------------------------
extern "C" void kernel_launch(void* const* d_in, const int* in_sizes, int n_in,
                              void* d_out, int out_size, void* d_ws, size_t ws_size,
                              hipStream_t stream) {
  const float* x    = (const float*)d_in[0];   // [4,2048,2048]
  const float* wqkv = (const float*)d_in[1];   // [6144,2048]
  const float* wout = (const float*)d_in[2];   // [2048,2048]
  float* out = (float*)d_out;                  // output | k | v

  char* ws = (char*)d_ws;
  size_t off = 0;
  auto take = [&](size_t bytes) -> char* {
    char* p = ws + off;
    off += (bytes + 255) & ~(size_t)255;
    return p;
  };
  // reg0: x(bf16) during GEMM1, then q(bf16,[B,H,T,hd]) after RoPE.
  unsigned short* reg0  = (unsigned short*)take(33554432);
  unsigned short* wqkvb = (unsigned short*)take(25165824);
  unsigned short* woutb = (unsigned short*)take(8388608);
  // reg1: qkv(bf16,[8192,6144]) during GEMM1/RoPE, then O(bf16,[8192,2048]).
  unsigned short* reg1  = (unsigned short*)take(100663296);
  unsigned short* kb    = (unsigned short*)take(33554432);
  unsigned short* vb    = (unsigned short*)take(33554432);

  float* kout = out + 16777216;   // after output [4,2048,2048]
  float* vout = out + 33554432;

  // 1) downconvert inputs to bf16
  cvt_f32_bf16<<<16777216 / 256, 256, 0, stream>>>(x, reg0, 16777216);
  cvt_f32_bf16<<<12582912 / 256, 256, 0, stream>>>(wqkv, wqkvb, 12582912);
  cvt_f32_bf16<<<4194304 / 256, 256, 0, stream>>>(wout, woutb, 4194304);

  // 2) qkv = x @ wqkv^T  : M=8192 N=6144 K=2048 -> bf16
  gemm_bf16_kernel<<<dim3(48, 64), 256, 0, stream>>>(reg0, wqkvb, nullptr, reg1,
                                                     8192, 6144, 2048);
  // 3) RoPE + scatter (also writes fp32 k, v outputs)
  rope_kernel<<<65536, 256, 0, stream>>>(reg1, reg0, kb, vb, kout, vout);

  // 4) flash attention -> O bf16 [8192,2048] (reuses reg1)
  flash_kernel<<<dim3(16, 16, 4), 256, 0, stream>>>(reg0, kb, vb, reg1);

  // 5) output = O @ w_out^T : M=8192 N=2048 K=2048 -> fp32 into d_out
  gemm_bf16_kernel<<<dim3(16, 64), 256, 0, stream>>>(reg1, woutb, out, nullptr,
                                                     8192, 2048, 2048);
}
